// RNNEncoder_65652870087061
// MI455X (gfx1250) — matrix operand order repaired and checked by reading, not compiled, not run on hardware
//
#include <hip/hip_runtime.h>

typedef unsigned short u16;
typedef __attribute__((ext_vector_type(16))) u16     u16x16;
typedef __attribute__((ext_vector_type(8)))  u16     u16x8;
typedef __attribute__((ext_vector_type(4)))  u16     u16x4;
typedef __attribute__((ext_vector_type(16))) __bf16  bf16x16;
typedef __attribute__((ext_vector_type(4)))  float   f32x4;
typedef __attribute__((ext_vector_type(8)))  float   v8f;

#define BATCH  4096
#define TSTEPS 35
#define UDIM   1024
#define K2     2048   // concatenated K: [x_t | h]

// float -> bf16 via the native convert (backend picks packed v_cvt on gfx1250)
__device__ __forceinline__ u16 f2bf(float f) {
  return __builtin_bit_cast(u16, (__bf16)f);
}

union Frag16 {                                // 8 VGPRs: one WMMA A/B operand
  u16x16  u;
  bf16x16 b;
  u16x8   h[2];
};

// ---- W (1024 x 2048 f32, row-major) -> bf16 copy, same layout ----
__global__ void convert_w_kernel(const float* __restrict__ W, u16* __restrict__ Wbf) {
  int i = blockIdx.x * blockDim.x + threadIdx.x;   // one f32x4 per thread, exact grid
  f32x4 v = ((const f32x4*)W)[i];
  u16x4 r;
  r[0] = f2bf(v[0]); r[1] = f2bf(v[1]); r[2] = f2bf(v[2]); r[3] = f2bf(v[3]);
  ((u16x4*)Wbf)[i] = r;
}

// ---- gather the strided x[:, t, :] slice into a contiguous bf16 buffer ----
// Bandwidth-bound: 16 MB f32 in, 8 MB bf16 out (stays hot in 192 MB L2).
__global__ void convert_xt_kernel(const float* __restrict__ x, int t,
                                  u16* __restrict__ xbf) {
  int tid = blockIdx.x * blockDim.x + threadIdx.x;     // BATCH*UDIM/4 threads, exact
  int m   = tid >> 8;                                  // row (UDIM/4 = 256 vec4/row)
  int c4  = tid & 255;
  f32x4 v = *(const f32x4*)(x + ((size_t)m * TSTEPS + t) * UDIM + c4 * 4);
  u16x4 r;
  r[0] = f2bf(v[0]); r[1] = f2bf(v[1]); r[2] = f2bf(v[2]); r[3] = f2bf(v[3]);
  *(u16x4*)(xbf + (size_t)m * UDIM + c4 * 4) = r;
}

// ---- one recurrence step: h_out = relu([x_t | h_in] @ W^T + b) ----
// Block: 128 threads (4 wave32s, 2x2), block tile 128x128, wave tile 64x64.
// 16 WMMAs per (4 A + 4 B) fragment loads -> 4x reuse both operands.
__global__ __launch_bounds__(128)
void rnn_step_kernel(const u16*  __restrict__ xbf,
                     const u16*  __restrict__ h_in,
                     const u16*  __restrict__ Wbf,
                     const float* __restrict__ bias,
                     u16*  __restrict__ h_out,
                     float* __restrict__ out_f32,
                     int use_h)
{
  const int lane = threadIdx.x & 31;
  const int wid  = threadIdx.x >> 5;         // 0..3
  const int wm   = wid >> 1;                 // 0..1 -> M sub-tile
  const int wn   = wid & 1;                  // 0..1 -> N sub-tile
  const int l16  = lane & 15;
  const int lh   = lane >> 4;
  const int klo  = lh * 8;                   // A-frag K phase per lane-half (ISA 16-bit A layout)

  const int mWave = blockIdx.y * 128 + wm * 64;
  const int nWave = blockIdx.x * 128 + wn * 64;

  v8f zero = {};
  v8f acc[4][4];
#pragma unroll
  for (int i = 0; i < 4; ++i)
#pragma unroll
    for (int j = 0; j < 4; ++j) acc[i][j] = zero;

  // B: column n = row n of W (row-major, stride K2); lane-half selects K 8-runs at 8h and 16+8h.
  const u16* wb = Wbf + (size_t)(nWave + l16) * K2 + lh * 8;
  const size_t arow = (size_t)(mWave + l16) * UDIM + klo;   // A row offset (frag i adds 16*UDIM)

  // ---- phase 0: K in [0,1024), A = x_t (bf16) ---- (direct param use keeps
  // address-space inference intact -> global_load, not flat_load)
#pragma unroll 2
  for (int kb = 0; kb < UDIM; kb += 32) {
    Frag16 a[4];
#pragma unroll
    for (int i = 0; i < 4; ++i) {
      const u16* p = xbf + arow + (size_t)i * 16 * UDIM + kb;
      a[i].h[0] = *(const u16x8*)(p);          // K run [kb+klo, +8)
      a[i].h[1] = *(const u16x8*)(p + 16);     // K run [kb+16+klo, +8)
    }
    Frag16 bf[4];
#pragma unroll
    for (int j = 0; j < 4; ++j) {
      bf[j].h[0] = *(const u16x8*)(wb + (size_t)j * 16 * K2 + kb);
      bf[j].h[1] = *(const u16x8*)(wb + (size_t)j * 16 * K2 + kb + 16);
    }
#pragma unroll
    for (int i = 0; i < 4; ++i)
#pragma unroll
      for (int j = 0; j < 4; ++j)
        acc[i][j] = __builtin_amdgcn_wmma_f32_16x16x32_bf16(
            false, a[i].b, false, bf[j].b, (short)0, acc[i][j], false, false);
  }

  // ---- phase 1: K in [1024,2048), A = h_{t-1} (bf16) ----
  if (use_h) {
#pragma unroll 2
    for (int kb = 0; kb < UDIM; kb += 32) {
      Frag16 a[4];
#pragma unroll
      for (int i = 0; i < 4; ++i) {
        const u16* p = h_in + arow + (size_t)i * 16 * UDIM + kb;
        a[i].h[0] = *(const u16x8*)(p);
        a[i].h[1] = *(const u16x8*)(p + 16);
      }
      Frag16 bf[4];
#pragma unroll
      for (int j = 0; j < 4; ++j) {
        bf[j].h[0] = *(const u16x8*)(wb + UDIM + (size_t)j * 16 * K2 + kb);
        bf[j].h[1] = *(const u16x8*)(wb + UDIM + (size_t)j * 16 * K2 + kb + 16);
      }
#pragma unroll
      for (int i = 0; i < 4; ++i)
#pragma unroll
        for (int j = 0; j < 4; ++j)
          acc[i][j] = __builtin_amdgcn_wmma_f32_16x16x32_bf16(
              false, a[i].b, false, bf[j].b, (short)0, acc[i][j], false, false);
    }
  }

  // ---- epilogue: + bias, ReLU, store bf16 h (and f32 final output) ----
  float bn[4];
#pragma unroll
  for (int j = 0; j < 4; ++j) bn[j] = bias[nWave + j * 16 + l16];

#pragma unroll
  for (int i = 0; i < 4; ++i) {
#pragma unroll
    for (int j = 0; j < 4; ++j) {
      const int mb = mWave + i * 16 + lh * 8;      // C/D layout: VGPR v -> M = v + 8*laneHalf
      const int n  = nWave + j * 16 + l16;
#pragma unroll
      for (int v = 0; v < 8; ++v) {
        float val = acc[i][j][v] + bn[j];
        val = val > 0.0f ? val : 0.0f;
        const size_t idx = (size_t)(mb + v) * UDIM + n;
        h_out[idx] = f2bf(val);
        if (out_f32) out_f32[idx] = val;
      }
    }
  }
}

extern "C" void kernel_launch(void* const* d_in, const int* in_sizes, int n_in,
                              void* d_out, int out_size, void* d_ws, size_t ws_size,
                              hipStream_t stream) {
  (void)in_sizes; (void)n_in; (void)out_size; (void)ws_size;
  const float* x = (const float*)d_in[0];
  const float* W = (const float*)d_in[1];
  const float* b = (const float*)d_in[2];
  float* out = (float*)d_out;

  char* ws  = (char*)d_ws;
  u16*  Wbf = (u16*)ws;                       //  4 MB bf16 W
  u16*  xbf = (u16*)(ws + (4u  << 20));       //  8 MB bf16 x_t slice
  u16*  hA  = (u16*)(ws + (12u << 20));       //  8 MB h ping
  u16*  hB  = (u16*)(ws + (20u << 20));       //  8 MB h pong

  convert_w_kernel<<<dim3((UDIM * K2 / 4) / 256), 256, 0, stream>>>(W, Wbf);

  u16* hin  = hA;
  u16* hout = hB;
  for (int t = 0; t < TSTEPS; ++t) {
    convert_xt_kernel<<<dim3((BATCH * UDIM / 4) / 256), 256, 0, stream>>>(x, t, xbf);
    rnn_step_kernel<<<dim3(UDIM / 128, BATCH / 128), 128, 0, stream>>>(
        xbf, hin, Wbf, b, hout,
        (t == TSTEPS - 1) ? out : nullptr,
        (t > 0) ? 1 : 0);
    u16* tmp = hin; hin = hout; hout = tmp;
  }
}
